// EmbNodeGNNGRU_43911745634887
// MI455X (gfx1250) — matrix operand order repaired and checked by reading, not compiled, not run on hardware
//
#include <hip/hip_runtime.h>
#include <hip/hip_bf16.h>
#include <math.h>

typedef __attribute__((ext_vector_type(16))) __bf16 v16bf;
typedef __attribute__((ext_vector_type(8)))  __bf16 v8bf;
typedef __attribute__((ext_vector_type(8)))  float  v8f;

#define N_NODES  131072
#define N_EDGES  2097152
#define N_GRAPHS 2048
#define SEQ_T    64
#define HID      128
#define IN_DIM   64
#define VOCAB    5000

__device__ __forceinline__ float gelu_exact(float v) {
    return 0.5f * v * (1.0f + erff(v * 0.70710678118654752f));
}

// ---------------------------------------------------------------- zero ----
__global__ void zero_kernel(float4* __restrict__ p, int n4) {
    int i = blockIdx.x * blockDim.x + threadIdx.x;
    if (i < n4) { float4 z = {0.f, 0.f, 0.f, 0.f}; p[i] = z; }
}

// ------------------------------------------------- embedding + L2 norm ----
// One wave per node: lanes 0-15 load the 64 raw features, lanes 16-31 load
// the 64-wide embedding row; reduce sum-of-squares across the wave32.
__global__ void embed_norm_kernel(const float* __restrict__ x,
                                  const float* __restrict__ emb,
                                  __bf16* __restrict__ h) {
    int wave = (blockIdx.x * blockDim.x + threadIdx.x) >> 5;
    int lane = threadIdx.x & 31;
    if (wave >= N_NODES) return;
    const float* xr = x + (size_t)wave * (IN_DIM + 1);   // row stride = 65
    int id = (int)xr[IN_DIM];
    id = id < 0 ? 0 : (id >= VOCAB ? VOCAB - 1 : id);
    float v0, v1, v2, v3;
    if (lane < 16) {
        const float* p = xr + lane * 4;
        v0 = p[0]; v1 = p[1]; v2 = p[2]; v3 = p[3];
    } else {
        const float* p = emb + (size_t)id * IN_DIM + (lane - 16) * 4;
        v0 = p[0]; v1 = p[1]; v2 = p[2]; v3 = p[3];
    }
    float ss = v0 * v0 + v1 * v1 + v2 * v2 + v3 * v3;
#pragma unroll
    for (int m = 16; m >= 1; m >>= 1) ss += __shfl_xor(ss, m, 32);
    float inv = 1.0f / sqrtf(ss);
    __bf16* hp = h + (size_t)wave * HID + lane * 4;
    hp[0] = (__bf16)(v0 * inv); hp[1] = (__bf16)(v1 * inv);
    hp[2] = (__bf16)(v2 * inv); hp[3] = (__bf16)(v3 * inv);
}

// ------------------------------------------------------- edge scatter ----
// One wave per edge; 32 lanes x 4 dims = 128 feature atomics. h (32 MB bf16)
// and agg (64 MB f32) are L2-resident (192 MB L2) so the f32 atomics stay
// on-chip. Edge indices are wave-uniform -> force scalar loads.
__global__ void scatter_kernel(const __bf16* __restrict__ h,
                               const int* __restrict__ srcI,
                               const int* __restrict__ dstI,
                               float* agg) {
    int wave = (blockIdx.x * blockDim.x + threadIdx.x) >> 5;
    int lane = threadIdx.x & 31;
    if (wave >= N_EDGES) return;
    int e = __builtin_amdgcn_readfirstlane(wave);
    int s = srcI[e], d = dstI[e];
    const __bf16* hp = h + (size_t)s * HID + lane * 4;
    float* ap = agg + (size_t)d * HID + lane * 4;
#pragma unroll
    for (int i = 0; i < 4; ++i) {
        float v = (float)hp[i];
        __hip_atomic_fetch_add(ap + i, v, __ATOMIC_RELAXED, __HIP_MEMORY_SCOPE_AGENT);
    }
}

// ---------------------------------- pack weights into WMMA B fragments ----
// Combined W = [Wa(K0 rows); Wb(K1 rows)], Nout cols. Fragment layout per the
// CDNA5 16-bit B 32x16 spec: value at ((tn*ktTotal+kt)*32+lane)*16+j is
// W[K = kt*32 + (j>>3)*16 + (lane>>4)*8 + (j&7)][N = tn*16 + (lane&15)].
__global__ void pack_weights_kernel(const float* __restrict__ Wa,
                                    const float* __restrict__ Wb,
                                    __bf16* __restrict__ Bp,
                                    int K0, int K1, int Nout) {
    int idx = blockIdx.x * blockDim.x + threadIdx.x;
    int ktTotal = (K0 + K1) >> 5;
    int nTiles  = Nout >> 4;
    int total   = nTiles * ktTotal * 32 * 16;
    if (idx >= total) return;
    int j    = idx & 15;
    int lane = (idx >> 4) & 31;
    int ft   = idx >> 9;             // tn*ktTotal + kt
    int kt   = ft % ktTotal;
    int tn   = ft / ktTotal;
    int K = kt * 32 + (j >> 3) * 16 + (lane >> 4) * 8 + (j & 7);
    int N = tn * 16 + (lane & 15);
    float v = (K < K0) ? Wa[(size_t)K * Nout + N]
                       : Wb[(size_t)(K - K0) * Nout + N];
    Bp[idx] = (__bf16)v;
}

// ------------------------------------- fused WMMA GEMM + bias + gelu ------
// out[M,NOUT] = gelu( A0[M,K0] @ W[0:K0] + A1[M,K1] @ W[K0:] + bias ).
// Shape is a template parameter so the whole K loop unrolls into straight-
// line WMMA with no per-iteration branches. One wave per 16x16 tile; block =
// 8 waves covering whole M-tile row(s), so the in-place layer update
// (A1 == outB) is race-free after the barrier. A-fragment per CDNA5 16-bit A
// 16x32 layout: lane holds row (lane&15), K chunks at half*8 and 16+half*8
// within each 32-K window.
template<int K0, int K1, int NOUT, bool OUTF32>
__global__ __launch_bounds__(256)
void gemm_bias_gelu_kernel(const float* __restrict__ A0,
                           const __bf16* A1,
                           const __bf16* __restrict__ Bp,
                           const float* __restrict__ bias,
                           float* outF, __bf16* outB) {
    constexpr int KT0 = K0 >> 5;
    constexpr int KT1 = K1 >> 5;
    constexpr int KT  = KT0 + KT1;
    constexpr int NT  = NOUT >> 4;

    int w    = threadIdx.x >> 5;
    int lane = threadIdx.x & 31;
    int gt    = blockIdx.x * 8 + w;
    int tileN = gt % NT;
    int tileM = gt / NT;
    int lh    = lane >> 4;
    int row   = tileM * 16 + (lane & 15);

    const __bf16* bBase = Bp + ((size_t)tileN * KT * 32 + lane) * 16;

    v8f acc = {0.f, 0.f, 0.f, 0.f, 0.f, 0.f, 0.f, 0.f};

    // ---- A0 (f32, convert to bf16 on the fly) ----
#pragma unroll
    for (int kt = 0; kt < KT0; ++kt) {
        const float* ar = A0 + (size_t)row * K0 + kt * 32 + lh * 8;
        float4 f0 = *(const float4*)(ar);
        float4 f1 = *(const float4*)(ar + 4);
        float4 f2 = *(const float4*)(ar + 16);
        float4 f3 = *(const float4*)(ar + 20);
        v16bf a;
        a[0]=(__bf16)f0.x;  a[1]=(__bf16)f0.y;  a[2]=(__bf16)f0.z;  a[3]=(__bf16)f0.w;
        a[4]=(__bf16)f1.x;  a[5]=(__bf16)f1.y;  a[6]=(__bf16)f1.z;  a[7]=(__bf16)f1.w;
        a[8]=(__bf16)f2.x;  a[9]=(__bf16)f2.y;  a[10]=(__bf16)f2.z; a[11]=(__bf16)f2.w;
        a[12]=(__bf16)f3.x; a[13]=(__bf16)f3.y; a[14]=(__bf16)f3.z; a[15]=(__bf16)f3.w;
        v16bf b = *(const v16bf*)(bBase + (size_t)kt * 32 * 16);
        acc = __builtin_amdgcn_wmma_f32_16x16x32_bf16(
                  false, a, false, b, (short)0, acc, false, false);
    }

    // ---- A1 (bf16, direct fragments) ----
#pragma unroll
    for (int kt = 0; kt < KT1; ++kt) {
        const __bf16* ar = A1 + (size_t)row * K1 + kt * 32 + lh * 8;
        v8bf c0 = *(const v8bf*)(ar);
        v8bf c1 = *(const v8bf*)(ar + 16);
        v16bf a;
#pragma unroll
        for (int i = 0; i < 8; ++i) { a[i] = c0[i]; a[8 + i] = c1[i]; }
        v16bf b = *(const v16bf*)(bBase + (size_t)(KT0 + kt) * 32 * 16);
        acc = __builtin_amdgcn_wmma_f32_16x16x32_bf16(
                  false, a, false, b, (short)0, acc, false, false);
    }

    int col  = tileN * 16 + (lane & 15);
    float bv = bias[col];
    __syncthreads();     // all reads of A1 (possibly == outB) done before stores
#pragma unroll
    for (int r = 0; r < 8; ++r) {
        int m   = tileM * 16 + lh * 8 + r;    // C/D layout: M = r + half*8
        float v = gelu_exact(acc[r] + bv);
        if (OUTF32) outF[(size_t)m * NOUT + col] = v;
        else        outB[(size_t)m * NOUT + col] = (__bf16)v;
    }
}

// ------------------------------------------- GRU input-gate precompute ----
__global__ void gi_kernel(const float* __restrict__ h64,
                          const float* __restrict__ Wih,
                          const float* __restrict__ bih,
                          float* __restrict__ gi) {
    int n = blockIdx.x * blockDim.x + threadIdx.x;
    if (n >= N_NODES) return;
    const float* xr = h64 + (size_t)n * IN_DIM;
    float a0 = bih[0], a1 = bih[1], a2 = bih[2];
    for (int k = 0; k < IN_DIM; ++k) {
        float xv = xr[k];
        a0 += xv * Wih[k];
        a1 += xv * Wih[IN_DIM + k];
        a2 += xv * Wih[2 * IN_DIM + k];
    }
    gi[(size_t)n * 3 + 0] = a0;
    gi[(size_t)n * 3 + 1] = a1;
    gi[(size_t)n * 3 + 2] = a2;
}

// ----------------------------------------------- GRU scan (hidden = 1) ----
__global__ void gru_scan_kernel(const float* __restrict__ gi,
                                const float* __restrict__ Whh,
                                const float* __restrict__ bhh,
                                const float* __restrict__ init,
                                float* __restrict__ out) {
    int b = blockIdx.x * blockDim.x + threadIdx.x;
    if (b >= N_GRAPHS) return;
    float hp = init[0];
    float w0 = Whh[0], w1 = Whh[1], w2 = Whh[2];
    float c0 = bhh[0], c1 = bhh[1], c2 = bhh[2];
    const float* g = gi + (size_t)b * SEQ_T * 3;
    float sum = 0.f, lastval = 0.f, pred0 = 0.f;
    int any = 0;
    for (int t = 0; t < SEQ_T; ++t) {
        float r = 1.f / (1.f + expf(-(g[t * 3 + 0] + w0 * hp + c0)));
        float z = 1.f / (1.f + expf(-(g[t * 3 + 1] + w1 * hp + c1)));
        float n = tanhf(g[t * 3 + 2] + r * (w2 * hp + c2));
        hp = (1.f - z) * n + z * hp;
        sum += hp;
        if (hp != 0.f) { lastval = hp; any = 1; }
        if (t == 0) pred0 = hp;
    }
    out[b] = (sum > 0.f && any) ? lastval : pred0;   // last-nonzero / fallback 0
}

// -------------------------------------------------------------------------
extern "C" void kernel_launch(void* const* d_in, const int* in_sizes, int n_in,
                              void* d_out, int out_size, void* d_ws, size_t ws_size,
                              hipStream_t stream) {
    (void)in_sizes; (void)n_in; (void)out_size; (void)ws_size;
    const float* x    = (const float*)d_in[0];
    const float* emb  = (const float*)d_in[1];
    const float* Wrel[3]  = {(const float*)d_in[2], (const float*)d_in[5], (const float*)d_in[8]};
    const float* brel[3]  = {(const float*)d_in[3], (const float*)d_in[6], (const float*)d_in[9]};
    const float* Wroot[3] = {(const float*)d_in[4], (const float*)d_in[7], (const float*)d_in[10]};
    const float* Wfc  = (const float*)d_in[11];
    const float* bfc  = (const float*)d_in[12];
    const float* Wih  = (const float*)d_in[13];
    const float* Whh  = (const float*)d_in[14];
    const float* bih  = (const float*)d_in[15];
    const float* bhh  = (const float*)d_in[16];
    const float* ih0  = (const float*)d_in[17];
    const int*   edge = (const int*)d_in[18];
    float* out = (float*)d_out;

    // workspace layout (bytes): h bf16 32MB | agg f32 64MB | h64 f32 32MB |
    // gi 1.5MB | packed weight fragments
    char* ws = (char*)d_ws;
    __bf16* h   = (__bf16*)(ws);
    float*  agg = (float*)(ws + (size_t)32 * 1024 * 1024);
    float*  h64 = (float*)(ws + (size_t)96 * 1024 * 1024);
    float*  gi  = (float*)(ws + (size_t)128 * 1024 * 1024);
    size_t bpOff = (size_t)130 * 1024 * 1024;
    __bf16* Bp0 = (__bf16*)(ws + bpOff);
    __bf16* Bp1 = (__bf16*)(ws + bpOff + 1 * 65536);
    __bf16* Bp2 = (__bf16*)(ws + bpOff + 2 * 65536);
    __bf16* BpF = (__bf16*)(ws + bpOff + 3 * 65536);
    __bf16* Bp[3] = {Bp0, Bp1, Bp2};

    // pack [W_rel; W_root] (256x128) per layer and W_fc (128x64)
    for (int l = 0; l < 3; ++l)
        pack_weights_kernel<<<128, 256, 0, stream>>>(Wrel[l], Wroot[l], Bp[l], 128, 128, 128);
    pack_weights_kernel<<<32, 256, 0, stream>>>(Wfc, Wfc, BpF, 128, 0, 64);

    embed_norm_kernel<<<N_NODES / 8, 256, 0, stream>>>(x, emb, h);

    const int* srcI = edge;
    const int* dstI = edge + N_EDGES;
    for (int l = 0; l < 3; ++l) {
        zero_kernel<<<(N_NODES * HID / 4) / 256, 256, 0, stream>>>((float4*)agg, N_NODES * HID / 4);
        scatter_kernel<<<N_EDGES / 8, 256, 0, stream>>>(h, srcI, dstI, agg);
        // in-place: block = 1 M-tile x 8 N-tiles, barrier before store
        gemm_bias_gelu_kernel<128, 128, 128, false><<<N_NODES / 16, 256, 0, stream>>>(
            agg, h, Bp[l], brel[l], nullptr, h);
    }
    // fc: [N,128] @ [128,64] -> f32 h64 (out-of-place)
    gemm_bias_gelu_kernel<0, 128, 64, true><<<(N_NODES / 16) * 4 / 8, 256, 0, stream>>>(
        nullptr, h, BpF, bfc, h64, nullptr);

    gi_kernel<<<N_NODES / 256, 256, 0, stream>>>(h64, Wih, bih, gi);
    gru_scan_kernel<<<N_GRAPHS / 256, 256, 0, stream>>>(gi, Whh, bhh, ih0, out);
}
